// SMARTRobertaClassificationModel_12403865551277
// MI455X (gfx1250) — matrix operand order repaired and checked by reading, not compiled, not run on hardware
//
#include <hip/hip_runtime.h>

#define B_  16
#define S_  512
#define D_  768
#define H_  12
#define DH_ 64

typedef __attribute__((ext_vector_type(16))) _Float16 v16h;
typedef __attribute__((ext_vector_type(8)))  _Float16 v8h;
typedef __attribute__((ext_vector_type(8)))  float    v8f;
typedef __attribute__((ext_vector_type(4)))  int      v4i;

#if defined(__gfx1250__)
#if __has_builtin(__builtin_amdgcn_global_load_async_to_lds_b128) && \
    __has_builtin(__builtin_amdgcn_s_wait_asynccnt)
#define USE_ASYNC 1
#endif
#endif

#define AS1 __attribute__((address_space(1)))
#define AS3 __attribute__((address_space(3)))

// 16-byte tile copy global->LDS: async DMA (ASYNCcnt) on CDNA5, sync fallback.
// Builtin prototype (from hipcc diagnostic): (int4 AS1*, int4 AS3*, imm, imm).
__device__ __forceinline__ void copy16(const _Float16* g, _Float16* l) {
#ifdef USE_ASYNC
    __builtin_amdgcn_global_load_async_to_lds_b128((AS1 v4i*)g, (AS3 v4i*)l, 0, 0);
#else
    *(v8h*)l = *(const v8h*)g;
#endif
}
__device__ __forceinline__ void async_wait() {
#ifdef USE_ASYNC
    __builtin_amdgcn_s_wait_asynccnt(0);
#endif
}

// A-fragment (16x32 f16, ISA 7.12.2): lane half h holds k runs [8h,8h+8) and
// [16+8h,16+8h+8)  -> two 16B loads + shuffle.
__device__ __forceinline__ v16h fragA(const _Float16* row, int hf) {
    v8h lo = *(const v8h*)(row + 8 * hf);
    v8h hi = *(const v8h*)(row + 16 + 8 * hf);
    return __builtin_shufflevector(lo, hi, 0, 1, 2, 3, 4, 5, 6, 7,
                                   8, 9, 10, 11, 12, 13, 14, 15);
}
// B-fragment (32x16 f16): same k order as A: lane half h holds k runs
// [8h,8h+8) and [16+8h,16+8h+8) -> two 16B loads + shuffle.
__device__ __forceinline__ v16h fragB(const _Float16* colrun, int hf) {
    v8h lo = *(const v8h*)(colrun + 8 * hf);
    v8h hi = *(const v8h*)(colrun + 16 + 8 * hf);
    return __builtin_shufflevector(lo, hi, 0, 1, 2, 3, 4, 5, 6, 7,
                                   8, 9, 10, 11, 12, 13, 14, 15);
}

// ---------------------------------------------------------------------------
// Kernel 0: f32 -> f16 conversion (once; feeds DMA-friendly f16 tiles).
// ---------------------------------------------------------------------------
__global__ void cvt_f16(const float* __restrict__ src, _Float16* __restrict__ dst,
                        int n) {
    for (int i = blockIdx.x * blockDim.x + threadIdx.x; i < n;
         i += gridDim.x * blockDim.x)
        dst[i] = (_Float16)src[i];
}

// ---------------------------------------------------------------------------
// Kernel 1: fused QKV GEMM.  Block = 128 thr (4 waves) does 64 rows x 64 cols
// of one of {q,k,v}; wave w owns rows [16w,16w+16) with 4 accumulators.
// Tiles are async-DMA'd to LDS as f16; V output is stored transposed
// [B,H,DH,S] so kernel 2's P@V B-fragments are contiguous.
// ---------------------------------------------------------------------------
__global__ __launch_bounds__(128)
void qkv_wmma(const _Float16* __restrict__ hsrc,   // [B*S, D] f16
              const _Float16* __restrict__ wq16,   // [D, D] f16 (rows = out col)
              const _Float16* __restrict__ wk16,
              const _Float16* __restrict__ wv16,
              const float* __restrict__ bq, const float* __restrict__ bk,
              const float* __restrict__ bv,
              _Float16* __restrict__ qo, _Float16* __restrict__ ko,
              _Float16* __restrict__ vo)
{
    __shared__ alignas(32) _Float16 As[64][32];   // 4 KB activations
    __shared__ alignas(32) _Float16 Bs[64][32];   // 4 KB weights

    const int m0  = blockIdx.x * 64;              // 128 blocks over B*S=8192
    const int yy  = blockIdx.y;                   // 0..35
    const int mat = yy / 12;                      // 0=q 1=k 2=v
    const int nc0 = (yy % 12) * 64;
    const _Float16* Wsel = (mat == 0) ? wq16 : (mat == 1) ? wk16 : wv16;
    const float*    bias = (mat == 0) ? bq : (mat == 1) ? bk : bv;
    _Float16*       outp = (mat == 0) ? qo : (mat == 1) ? ko : vo;

    const int t = threadIdx.x, wave = t >> 5, lane = t & 31;
    const int hf = lane >> 4, li = lane & 15;

    v8f acc[4] = {{}, {}, {}, {}};
    for (int kc = 0; kc < D_; kc += 32) {
        // stage 64x32 f16 A and B tiles: 256 16B slots each, 2 per thread
        #pragma unroll
        for (int s = t; s < 256; s += 128) {
            int row = s >> 2, c8 = (s & 3) * 8;
            copy16(hsrc + (size_t)(m0 + row) * D_ + kc + c8, &As[0][0] + s * 8);
            copy16(Wsel + (size_t)(nc0 + row) * D_ + kc + c8, &Bs[0][0] + s * 8);
        }
        async_wait();
        __syncthreads();

        v16h a = fragA(&As[wave * 16 + li][0], hf);
        #pragma unroll
        for (int j = 0; j < 4; ++j) {
            v16h b = fragB(&Bs[j * 16 + li][0], hf);
            acc[j] = __builtin_amdgcn_wmma_f32_16x16x32_f16(
                false, a, false, b, (short)0, acc[j], false, false);
        }
        __syncthreads();
    }

    // epilogue: bias + layout.  q/k -> [B,H,S,DH]; v -> [B,H,DH,S] (transposed)
    #pragma unroll
    for (int j = 0; j < 4; ++j) {
        const int ncol = nc0 + j * 16 + li;
        const int head = ncol / DH_, dh = ncol % DH_;
        const float bb = bias[ncol];
        #pragma unroll
        for (int r = 0; r < 8; ++r) {
            int m = m0 + wave * 16 + r + 8 * hf;
            int bi = m / S_, s = m % S_;
            size_t o = (mat == 2)
                ? ((((size_t)bi * H_ + head) * DH_ + dh) * S_ + s)
                : ((((size_t)bi * H_ + head) * S_ + s) * DH_ + dh);
            outp[o] = (_Float16)(acc[j][r] + bb);
        }
    }
}

// ---------------------------------------------------------------------------
// Kernel 2: fused scores -> sparsemax -> context for one (b,h) x 16 queries.
// Scores live in LDS only (never HBM).
// ---------------------------------------------------------------------------
__global__ __launch_bounds__(128)
void attn_sparsemax(const _Float16* __restrict__ qg,   // [B,H,S,DH]
                    const _Float16* __restrict__ kg,   // [B,H,S,DH]
                    const _Float16* __restrict__ vT,   // [B,H,DH,S]
                    const float*    __restrict__ mask, // [B,1,1,S]
                    float*          __restrict__ out)  // [B,S,D]
{
    __shared__ float             sc[16][S_];           // 32 KB scores
    __shared__ alignas(32) _Float16 pr[16][S_];        // 16 KB probs

    const int qt = blockIdx.x, bh = blockIdx.y;
    const int bi = bh / H_, head = bh % H_;
    const size_t base = (size_t)bh * S_ * DH_;

    const int t = threadIdx.x, wave = t >> 5, lane = t & 31;
    const int hf = lane >> 4, li = lane & 15;
    const int q0 = qt * 16;

    // q fragments: K=64 -> two 32-chunks, vector loads
    v16h aq[2];
    #pragma unroll
    for (int c = 0; c < 2; ++c)
        aq[c] = fragA(qg + base + (size_t)(q0 + li) * DH_ + c * 32, hf);

    // --- phase 1: scores = q k^T / 8 + mask ---------------------------------
    for (int kt = wave; kt < S_ / 16; kt += 4) {
        const int key0 = kt * 16;
        v8f c = {};
        #pragma unroll
        for (int cc = 0; cc < 2; ++cc) {
            v16h b = fragB(kg + base + (size_t)(key0 + li) * DH_ + cc * 32, hf);
            c = __builtin_amdgcn_wmma_f32_16x16x32_f16(
                false, aq[cc], false, b, (short)0, c, false, false);
        }
        const float mk = mask[(size_t)bi * S_ + key0 + li];
        #pragma unroll
        for (int r = 0; r < 8; ++r)
            sc[r + 8 * hf][key0 + li] = c[r] * 0.125f + mk;
    }
    __syncthreads();

    // --- phase 2: sparsemax per row (registers + wave32 reductions) ---------
    for (int rr = 0; rr < 4; ++rr) {
        const int row = wave * 4 + rr;
        float z[16];
        #pragma unroll
        for (int i = 0; i < 16; ++i) z[i] = sc[row][lane + 32 * i];

        float mx = z[0];
        #pragma unroll
        for (int i = 1; i < 16; ++i) mx = fmaxf(mx, z[i]);
        #pragma unroll
        for (int o = 16; o > 0; o >>= 1) mx = fmaxf(mx, __shfl_xor(mx, o, 32));

        // g(tau)=sum(max(z-tau,0)) monotone; g(mx-1)>=1>=g(mx): bisect
        float lo = mx - 1.0f, hi = mx;
        for (int it = 0; it < 30; ++it) {
            float tau = 0.5f * (lo + hi);
            float s = 0.0f;
            #pragma unroll
            for (int i = 0; i < 16; ++i) s += fmaxf(z[i] - tau, 0.0f);
            #pragma unroll
            for (int o = 16; o > 0; o >>= 1) s += __shfl_xor(s, o, 32);
            if (s >= 1.0f) lo = tau; else hi = tau;
        }
        const float tau = 0.5f * (lo + hi);

        // exact refinement: tau* = (sum(support) - 1) / |support|
        float cnt = 0.0f, ssum = 0.0f;
        #pragma unroll
        for (int i = 0; i < 16; ++i)
            if (z[i] > tau) { cnt += 1.0f; ssum += z[i]; }
        #pragma unroll
        for (int o = 16; o > 0; o >>= 1) {
            cnt  += __shfl_xor(cnt,  o, 32);
            ssum += __shfl_xor(ssum, o, 32);
        }
        const float taux = (ssum - 1.0f) / fmaxf(cnt, 1.0f);
        #pragma unroll
        for (int i = 0; i < 16; ++i)
            pr[row][lane + 32 * i] = (_Float16)fmaxf(z[i] - taux, 0.0f);
    }
    __syncthreads();

    // --- phase 3: ctx = probs @ v; wave owns dh cols [16*wave, 16*wave+16) --
    {
        const int n0 = wave * 16;
        const size_t vbase = base + (size_t)(n0 + li) * S_;
        v8f c = {};
        for (int kc = 0; kc < S_; kc += 32) {
            v16h a = fragA(&pr[li][kc], hf);
            v16h b = fragB(vT + vbase + kc, hf);
            c = __builtin_amdgcn_wmma_f32_16x16x32_f16(
                false, a, false, b, (short)0, c, false, false);
        }
        #pragma unroll
        for (int r = 0; r < 8; ++r) {
            int m = q0 + r + 8 * hf;
            out[((size_t)bi * S_ + m) * D_ + head * DH_ + n0 + li] = c[r];
        }
    }
}

// ---------------------------------------------------------------------------
extern "C" void kernel_launch(void* const* d_in, const int* in_sizes, int n_in,
                              void* d_out, int out_size, void* d_ws, size_t ws_size,
                              hipStream_t stream)
{
    const float* hidden = (const float*)d_in[0];
    const float* mask   = (const float*)d_in[1];
    const float* Wq     = (const float*)d_in[2];
    const float* bq     = (const float*)d_in[3];
    const float* Wk     = (const float*)d_in[4];
    const float* bk     = (const float*)d_in[5];
    const float* Wv     = (const float*)d_in[6];
    const float* bv     = (const float*)d_in[7];
    float*       out    = (float*)d_out;

    const size_t nAct = (size_t)B_ * S_ * D_;      // 6,291,456
    const size_t nW   = (size_t)D_ * D_;           //   589,824

    _Float16* hs   = (_Float16*)d_ws;
    _Float16* wq16 = hs + nAct;
    _Float16* wk16 = wq16 + nW;
    _Float16* wv16 = wk16 + nW;
    _Float16* qg   = wv16 + nW;
    _Float16* kg   = qg + nAct;
    _Float16* vgT  = kg + nAct;

    cvt_f16<<<2048, 256, 0, stream>>>(hidden, hs, (int)nAct);
    cvt_f16<<<1024, 256, 0, stream>>>(Wq, wq16, (int)nW);
    cvt_f16<<<1024, 256, 0, stream>>>(Wk, wk16, (int)nW);
    cvt_f16<<<1024, 256, 0, stream>>>(Wv, wv16, (int)nW);

    dim3 blk(128);
    dim3 g1(128, 36);                 // 64-row x 64-col tiles, 3 matrices
    qkv_wmma<<<g1, blk, 0, stream>>>(hs, wq16, wk16, wv16, bq, bk, bv,
                                     qg, kg, vgT);

    dim3 g2(S_ / 16, B_ * H_);
    attn_sparsemax<<<g2, blk, 0, stream>>>(qg, kg, vgT, mask, out);
}
